// Model_39960375722255
// MI455X (gfx1250) — matrix operand order repaired and checked by reading, not compiled, not run on hardware
//
#include <hip/hip_runtime.h>

#define NCELLS 20000
#define NGENES 3000
#define HIDDIM 512
#define OUTDIM 128
#define ECELL  320000
#define EGENE  48000

typedef __attribute__((ext_vector_type(8)))  float  v8f;
typedef __attribute__((ext_vector_type(16))) __bf16 v16bf;

union FragAB { v16bf v; unsigned int u[8]; uint4 q[2]; };
union FragC  { v8f  v; float f[8]; };

__device__ __forceinline__ unsigned short f2bf(float x) {
  unsigned int u = __float_as_uint(x);
  u += 0x7FFFu + ((u >> 16) & 1u);   // round-to-nearest-even
  return (unsigned short)(u >> 16);
}

// ---------------------------------------------------------------- utilities
__global__ void k_zero_f32(float* __restrict__ p, int n) {
  int i = blockIdx.x * blockDim.x + threadIdx.x;
  if (i < n) p[i] = 0.0f;
}

__global__ void k_cast_bf16(const float* __restrict__ s, unsigned short* __restrict__ d, int n) {
  int i = blockIdx.x * blockDim.x + threadIdx.x;
  if (i < n) d[i] = f2bf(s[i]);
}

// src [R x C] f32  ->  dst [C x R] bf16
__global__ void k_transpose_cast(const float* __restrict__ src, unsigned short* __restrict__ dst,
                                 int R, int C) {
  __shared__ float tile[32][33];
  int c0 = blockIdx.x * 32, r0 = blockIdx.y * 32;
  #pragma unroll
  for (int j = 0; j < 4; ++j) {
    int r = r0 + threadIdx.y + j * 8;
    int c = c0 + threadIdx.x;
    tile[threadIdx.y + j * 8][threadIdx.x] = (r < R && c < C) ? src[(size_t)r * C + c] : 0.0f;
  }
  __syncthreads();
  #pragma unroll
  for (int j = 0; j < 4; ++j) {
    int cT = c0 + threadIdx.y + j * 8;   // src column = dst row
    int rT = r0 + threadIdx.x;           // src row    = dst col
    if (cT < C && rT < R)
      dst[(size_t)cT * R + rT] = f2bf(tile[threadIdx.x][threadIdx.y + j * 8]);
  }
}

__global__ void k_deg_count(const int* __restrict__ dst, int E, float* __restrict__ deg) {
  int e = blockIdx.x * blockDim.x + threadIdx.x;
  if (e < E) atomicAdd(&deg[dst[e]], 1.0f);
}

__global__ void k_finalize_dinv(float* __restrict__ deg, int n) {
  int i = blockIdx.x * blockDim.x + threadIdx.x;
  if (i < n) deg[i] = rsqrtf(deg[i] + 1.0f);   // +1 self loop; >=1 guaranteed
}

// per edge: agg[dst] += H[src] * dinv[src]*dinv[dst]
__global__ void k_scatter_agg(const int* __restrict__ src, const int* __restrict__ dst,
                              const float* __restrict__ H, const float* __restrict__ dinv,
                              float* __restrict__ agg, int F) {
  int e = blockIdx.x;
  int f = blockIdx.y * blockDim.x + threadIdx.x;
  int s = src[e], d = dst[e];
  float w = dinv[s] * dinv[d];
  atomicAdd(&agg[(size_t)d * F + f], H[(size_t)s * F + f] * w);
}

// out = act(agg + H*dinv^2 + bias); ACT: 0 none, 1 relu, 2 leaky
template<int ACT>
__global__ void k_agg_epilogue(const float* __restrict__ agg, const float* __restrict__ H,
                               const float* __restrict__ dinv, const float* __restrict__ bias,
                               float* __restrict__ outf, unsigned short* __restrict__ outbf,
                               int Nn, int F) {
  int idx = blockIdx.x * blockDim.x + threadIdx.x;
  if (idx >= Nn * F) return;
  int i = idx / F, f = idx - i * F;
  float di = dinv[i];
  float v = agg[idx] + H[idx] * di * di + bias[f];
  if (ACT == 1) v = fmaxf(v, 0.0f);
  else if (ACT == 2) v = (v >= 0.0f) ? v : 0.01f * v;
  if (outf)  outf[idx]  = v;
  if (outbf) outbf[idx] = f2bf(v);
}

// in-place row L2 norm, one wave32 per row, F multiple of 128
__global__ void k_l2norm_rows(float* __restrict__ x, int Nrows, int F) {
  int wave = (blockIdx.x * blockDim.x + threadIdx.x) >> 5;
  int lane = threadIdx.x & 31;
  if (wave >= Nrows) return;
  float4* row = reinterpret_cast<float4*>(x + (size_t)wave * F);
  int nq = F >> 2;
  float s = 0.0f;
  for (int i = lane; i < nq; i += 32) {
    float4 v = row[i];
    s += v.x * v.x + v.y * v.y + v.z * v.z + v.w * v.w;
  }
  #pragma unroll
  for (int o = 16; o > 0; o >>= 1) s += __shfl_xor(s, o, 32);
  float sc = 1.0f / fmaxf(sqrtf(s), 1e-12f);
  for (int i = lane; i < nq; i += 32) {
    float4 v = row[i];
    v.x *= sc; v.y *= sc; v.z *= sc; v.w *= sc;
    row[i] = v;
  }
}

// bpxg[t] = sum_g bp[g] * xg[g*128 + t]
__global__ void k_bpxg(const float* __restrict__ bp, const float* __restrict__ xg,
                       float* __restrict__ out) {
  int t = threadIdx.x;  // 128 threads
  float s = 0.0f;
  for (int g = 0; g < NGENES; ++g) s += bp[g] * xg[(size_t)g * OUTDIM + t];
  out[t] = s;
}

// ---------------------------------------------------------------- WMMA GEMM
// Both operands are stored K-contiguous: A is [M x K] bf16, Bt is [N x K] bf16
// (i.e. B pre-transposed). Every WMMA fragment is then 2 contiguous 16-byte
// LDS reads (ds_load_b128) with zero repacking VALU.
#define BM  128
#define BN  128
#define BK  32
#define GLD 40     // ushorts per LDS row (32 + 8 pad); 80B rows keep 16B alignment

// C[M,N] = act(A @ Bt^T + bias); optional bf16 copy, optionally transposed [N x M]
template<int ACT, int CBT>
__global__ __launch_bounds__(256)
void k_gemm_bf16(const unsigned short* __restrict__ A, const unsigned short* __restrict__ Bt,
                 const float* __restrict__ bias, float* __restrict__ C,
                 unsigned short* __restrict__ Cbf, int M, int N, int K) {
  __shared__ unsigned short Alds[BM * GLD];
  __shared__ unsigned short Blds[BN * GLD];

  const int tid  = threadIdx.x;
  const int lane = tid & 31;
  const int wave = tid >> 5;
  const int waveM = wave & 3;         // 4 waves along M (32 rows each)
  const int waveN = wave >> 2;        // 2 waves along N (64 cols each)
  const int bM = blockIdx.y * BM;
  const int bN = blockIdx.x * BN;

  FragC acc[2][4];
  #pragma unroll
  for (int i = 0; i < 2; ++i)
    #pragma unroll
    for (int j = 0; j < 4; ++j) acc[i][j].v = (v8f){0,0,0,0,0,0,0,0};

  const int ldRow = tid >> 2;         // 0..63
  const int ldCol = (tid & 3) * 8;    // 0,8,16,24
  const int mrow  = lane & 15;
  const int half  = lane >> 4;

  const int KT = (K + BK - 1) / BK;
  for (int kt = 0; kt < KT; ++kt) {
    const int k0 = kt * BK;
    #pragma unroll
    for (int p = 0; p < 2; ++p) {     // A tile: 128 rows x 32 k
      int row = ldRow + p * 64;
      int gm = bM + row, gk = k0 + ldCol;
      uint4 d = make_uint4(0, 0, 0, 0);
      if (gm < M && gk + 7 < K)       // K is a multiple of 8 in all our calls
        d = *reinterpret_cast<const uint4*>(&A[(size_t)gm * K + gk]);
      *reinterpret_cast<uint4*>(&Alds[row * GLD + ldCol]) = d;
    }
    #pragma unroll
    for (int p = 0; p < 2; ++p) {     // B tile: 128 n-rows x 32 k
      int row = ldRow + p * 64;
      int gn = bN + row, gk = k0 + ldCol;   // gn < N by grid construction
      uint4 d = make_uint4(0, 0, 0, 0);
      if (gk + 7 < K)
        d = *reinterpret_cast<const uint4*>(&Bt[(size_t)gn * K + gk]);
      *reinterpret_cast<uint4*>(&Blds[row * GLD + ldCol]) = d;
    }
    __syncthreads();

    FragAB a[2], b[4];
    #pragma unroll
    for (int mt = 0; mt < 2; ++mt) {  // A frag: ISA 16x32 16-bit layout
      int abase = (waveM * 32 + mt * 16 + mrow) * GLD + half * 8;
      a[mt].q[0] = *reinterpret_cast<const uint4*>(&Alds[abase]);
      a[mt].q[1] = *reinterpret_cast<const uint4*>(&Alds[abase + 16]);
    }
    #pragma unroll
    for (int nt = 0; nt < 4; ++nt) {  // B frag: ISA 16x32 16-bit layout
      int bbase = (waveN * 64 + nt * 16 + mrow) * GLD + half * 8;
      b[nt].q[0] = *reinterpret_cast<const uint4*>(&Blds[bbase]);
      b[nt].q[1] = *reinterpret_cast<const uint4*>(&Blds[bbase + 16]);
    }
    #pragma unroll
    for (int mt = 0; mt < 2; ++mt)
      #pragma unroll
      for (int nt = 0; nt < 4; ++nt)
        acc[mt][nt].v = __builtin_amdgcn_wmma_f32_16x16x32_bf16(
            false, a[mt].v, false, b[nt].v, (short)0, acc[mt][nt].v, false, false);
    __syncthreads();
  }

  const int mrow8 = half * 8;
  #pragma unroll
  for (int mt = 0; mt < 2; ++mt)
    #pragma unroll
    for (int r = 0; r < 8; ++r) {
      int gm = bM + waveM * 32 + mt * 16 + mrow8 + r;
      if (gm >= M) continue;
      #pragma unroll
      for (int nt = 0; nt < 4; ++nt) {
        int gn = bN + waveN * 64 + nt * 16 + mrow;
        float v = acc[mt][nt].f[r];
        if (bias) v += bias[gn];
        if (ACT == 1) v = fmaxf(v, 0.0f);
        else if (ACT == 2) v = (v >= 0.0f) ? v : 0.01f * v;
        if (C)   C[(size_t)gm * N + gn] = v;
        if (Cbf) {
          size_t ib = CBT ? ((size_t)gn * M + gm) : ((size_t)gm * N + gn);
          Cbf[ib] = f2bf(v);
        }
      }
    }
}

// Z[M,128] = l2norm_rows(A[M,K] @ Bt[128,K]^T + bias)  (grid.x must be 1)
__global__ __launch_bounds__(256)
void k_gemm_z_bf16(const unsigned short* __restrict__ A, const unsigned short* __restrict__ Bt,
                   const float* __restrict__ bias, float* __restrict__ Z, int M, int K) {
  __shared__ unsigned short Alds[BM * GLD];
  __shared__ unsigned short Blds[BN * GLD];
  __shared__ float sumsq[BM];
  const int N = 128;

  const int tid  = threadIdx.x;
  const int lane = tid & 31;
  const int wave = tid >> 5;
  const int waveM = wave & 3;
  const int waveN = wave >> 2;
  const int bM = blockIdx.y * BM;

  FragC acc[2][4];
  #pragma unroll
  for (int i = 0; i < 2; ++i)
    #pragma unroll
    for (int j = 0; j < 4; ++j) acc[i][j].v = (v8f){0,0,0,0,0,0,0,0};

  const int ldRow = tid >> 2, ldCol = (tid & 3) * 8;
  const int mrow  = lane & 15, half = lane >> 4;

  const int KT = (K + BK - 1) / BK;
  for (int kt = 0; kt < KT; ++kt) {
    const int k0 = kt * BK;
    #pragma unroll
    for (int p = 0; p < 2; ++p) {
      int row = ldRow + p * 64;
      int gm = bM + row, gk = k0 + ldCol;
      uint4 d = make_uint4(0, 0, 0, 0);
      if (gm < M && gk + 7 < K)
        d = *reinterpret_cast<const uint4*>(&A[(size_t)gm * K + gk]);
      *reinterpret_cast<uint4*>(&Alds[row * GLD + ldCol]) = d;
    }
    #pragma unroll
    for (int p = 0; p < 2; ++p) {
      int row = ldRow + p * 64;
      int gk = k0 + ldCol;
      uint4 d = make_uint4(0, 0, 0, 0);
      if (gk + 7 < K)
        d = *reinterpret_cast<const uint4*>(&Bt[(size_t)row * K + gk]);
      *reinterpret_cast<uint4*>(&Blds[row * GLD + ldCol]) = d;
    }
    __syncthreads();

    FragAB a[2], b[4];
    #pragma unroll
    for (int mt = 0; mt < 2; ++mt) {
      int abase = (waveM * 32 + mt * 16 + mrow) * GLD + half * 8;
      a[mt].q[0] = *reinterpret_cast<const uint4*>(&Alds[abase]);
      a[mt].q[1] = *reinterpret_cast<const uint4*>(&Alds[abase + 16]);
    }
    #pragma unroll
    for (int nt = 0; nt < 4; ++nt) {
      int bbase = (waveN * 64 + nt * 16 + mrow) * GLD + half * 8;
      b[nt].q[0] = *reinterpret_cast<const uint4*>(&Blds[bbase]);
      b[nt].q[1] = *reinterpret_cast<const uint4*>(&Blds[bbase + 16]);
    }
    #pragma unroll
    for (int mt = 0; mt < 2; ++mt)
      #pragma unroll
      for (int nt = 0; nt < 4; ++nt)
        acc[mt][nt].v = __builtin_amdgcn_wmma_f32_16x16x32_bf16(
            false, a[mt].v, false, b[nt].v, (short)0, acc[mt][nt].v, false, false);
    __syncthreads();
  }

  // fused bias + row L2-norm epilogue (BN == N == 128 -> full rows in block)
  if (tid < BM) sumsq[tid] = 0.0f;
  __syncthreads();
  const int mrow8 = half * 8;
  #pragma unroll
  for (int mt = 0; mt < 2; ++mt)
    #pragma unroll
    for (int r = 0; r < 8; ++r) {
      int lm = waveM * 32 + mt * 16 + mrow8 + r;
      float p = 0.0f;
      #pragma unroll
      for (int nt = 0; nt < 4; ++nt) {
        int gn = waveN * 64 + nt * 16 + mrow;
        float v = acc[mt][nt].f[r] + bias[gn];
        acc[mt][nt].f[r] = v;
        p += v * v;
      }
      atomicAdd(&sumsq[lm], p);   // ds_add_f32 across the 2 N-waves
    }
  __syncthreads();
  #pragma unroll
  for (int mt = 0; mt < 2; ++mt)
    #pragma unroll
    for (int r = 0; r < 8; ++r) {
      int lm = waveM * 32 + mt * 16 + mrow8 + r;
      int gm = bM + lm;
      if (gm >= M) continue;
      float sc = 1.0f / fmaxf(sqrtf(sumsq[lm]), 1e-12f);
      #pragma unroll
      for (int nt = 0; nt < 4; ++nt) {
        int gn = waveN * 64 + nt * 16 + mrow;
        Z[(size_t)gm * N + gn] = acc[mt][nt].f[r] * sc;
      }
    }
}

// ---------------------------------------------------------------- launcher
extern "C" void kernel_launch(void* const* d_in, const int* in_sizes, int n_in,
                              void* d_out, int out_size, void* d_ws, size_t ws_size,
                              hipStream_t stream) {
  (void)in_sizes; (void)n_in; (void)out_size; (void)ws_size;
  const float* x0 = (const float*)d_in[0];
  const float* x1 = (const float*)d_in[1];
  const int*   ei  = (const int*)d_in[2];     // [2, ECELL]
  const int*   eig = (const int*)d_in[3];     // [2, EGENE]
  const float* Wm = (const float*)d_in[4];
  const float* bm = (const float*)d_in[5];
  const float* W1 = (const float*)d_in[6];
  const float* b1 = (const float*)d_in[7];
  const float* W2 = (const float*)d_in[8];
  const float* b2 = (const float*)d_in[9];
  const float* W3 = (const float*)d_in[10];
  const float* b3 = (const float*)d_in[11];
  const float* Wp = (const float*)d_in[12];
  const float* bp = (const float*)d_in[13];
  const float* Wg = (const float*)d_in[14];
  const float* bg = (const float*)d_in[15];

  float* emb1 = (float*)d_out;
  float* emb2 = emb1 + (size_t)NCELLS * HIDDIM;
  float* z1   = emb2 + (size_t)NCELLS * HIDDIM;
  float* z2   = z1   + (size_t)NCELLS * OUTDIM;

  char* ws = (char*)d_ws;
  size_t off = 0;
  auto wsa = [&](size_t bytes) { void* p = ws + off; off += (bytes + 255) & ~(size_t)255; return p; };
  unsigned short* x0bf    = (unsigned short*)wsa((size_t)NCELLS * NGENES * 2);  // [20000 x 3000]
  unsigned short* x1bf    = (unsigned short*)wsa((size_t)NCELLS * NGENES * 2);
  unsigned short* x0Tbf   = (unsigned short*)wsa((size_t)NGENES * NCELLS * 2);  // [3000 x 20000]
  unsigned short* WmTbf   = (unsigned short*)wsa((size_t)HIDDIM * NGENES * 2);  // [512 x 3000]
  unsigned short* W1Tbf   = (unsigned short*)wsa((size_t)HIDDIM * NGENES * 2);
  unsigned short* W2Tbf   = (unsigned short*)wsa((size_t)HIDDIM * HIDDIM * 2);
  unsigned short* W3Tbf   = (unsigned short*)wsa((size_t)HIDDIM * HIDDIM * 2);
  unsigned short* Wpbf    = (unsigned short*)wsa((size_t)HIDDIM * NGENES * 2);  // [512 x 3000] (A op)
  unsigned short* WgTbf   = (unsigned short*)wsa((size_t)OUTDIM * NCELLS * 2);  // [128 x 20000]
  float*          Hbuf    = (float*)wsa((size_t)NCELLS * HIDDIM * 4);
  float*          AGG     = (float*)wsa((size_t)NCELLS * HIDDIM * 4);
  unsigned short* h1bf    = (unsigned short*)wsa((size_t)NCELLS * HIDDIM * 2);
  unsigned short* h2bf    = (unsigned short*)wsa((size_t)NCELLS * HIDDIM * 2);
  unsigned short* h2abf   = (unsigned short*)wsa((size_t)NCELLS * HIDDIM * 2);
  unsigned short* h2bbf   = (unsigned short*)wsa((size_t)NCELLS * HIDDIM * 2);
  float*          dinvc   = (float*)wsa((size_t)NCELLS * 4);
  float*          dinvg   = (float*)wsa((size_t)NGENES * 4);
  float*          xgf     = (float*)wsa((size_t)NGENES * OUTDIM * 4);           // [3000 x 128]
  unsigned short* xgTbf   = (unsigned short*)wsa((size_t)OUTDIM * NGENES * 2);  // [128 x 3000]
  unsigned short* WpxgTbf = (unsigned short*)wsa((size_t)OUTDIM * HIDDIM * 2);  // [128 x 512]
  float*          bpxg    = (float*)wsa((size_t)OUTDIM * 4);

  auto cdiv = [](int a, int b) { return (a + b - 1) / b; };

  // bf16 prep: A operands row-major, B operands transposed ([N x K])
  k_cast_bf16<<<cdiv(NCELLS * NGENES, 256), 256, 0, stream>>>(x0, x0bf, NCELLS * NGENES);
  k_cast_bf16<<<cdiv(NCELLS * NGENES, 256), 256, 0, stream>>>(x1, x1bf, NCELLS * NGENES);
  k_cast_bf16<<<cdiv(HIDDIM * NGENES, 256), 256, 0, stream>>>(Wp, Wpbf, HIDDIM * NGENES);
  k_transpose_cast<<<dim3(cdiv(NGENES, 32), cdiv(NCELLS, 32)), dim3(32, 8), 0, stream>>>(
      x0, x0Tbf, NCELLS, NGENES);
  k_transpose_cast<<<dim3(cdiv(HIDDIM, 32), cdiv(NGENES, 32)), dim3(32, 8), 0, stream>>>(
      Wm, WmTbf, NGENES, HIDDIM);
  k_transpose_cast<<<dim3(cdiv(HIDDIM, 32), cdiv(NGENES, 32)), dim3(32, 8), 0, stream>>>(
      W1, W1Tbf, NGENES, HIDDIM);
  k_transpose_cast<<<dim3(cdiv(HIDDIM, 32), cdiv(HIDDIM, 32)), dim3(32, 8), 0, stream>>>(
      W2, W2Tbf, HIDDIM, HIDDIM);
  k_transpose_cast<<<dim3(cdiv(HIDDIM, 32), cdiv(HIDDIM, 32)), dim3(32, 8), 0, stream>>>(
      W3, W3Tbf, HIDDIM, HIDDIM);
  k_transpose_cast<<<dim3(cdiv(OUTDIM, 32), cdiv(NCELLS, 32)), dim3(32, 8), 0, stream>>>(
      Wg, WgTbf, NCELLS, OUTDIM);

  // degree -> dinv (stored in place)
  k_zero_f32<<<cdiv(NCELLS, 256), 256, 0, stream>>>(dinvc, NCELLS);
  k_deg_count<<<cdiv(ECELL, 256), 256, 0, stream>>>(ei + ECELL, ECELL, dinvc);
  k_finalize_dinv<<<cdiv(NCELLS, 256), 256, 0, stream>>>(dinvc, NCELLS);
  k_zero_f32<<<cdiv(NGENES, 256), 256, 0, stream>>>(dinvg, NGENES);
  k_deg_count<<<cdiv(EGENE, 256), 256, 0, stream>>>(eig + EGENE, EGENE, dinvg);
  k_finalize_dinv<<<cdiv(NGENES, 256), 256, 0, stream>>>(dinvg, NGENES);

  const dim3 gemmNH(HIDDIM / BN, cdiv(NCELLS, BM));   // (4, 157)

  // GCN layer 1: H = x1@W1 ; aggregate ; leaky -> h2abf
  k_gemm_bf16<0,0><<<gemmNH, 256, 0, stream>>>(x1bf, W1Tbf, nullptr, Hbuf, nullptr,
                                               NCELLS, HIDDIM, NGENES);
  k_zero_f32<<<cdiv(NCELLS * HIDDIM, 256), 256, 0, stream>>>(AGG, NCELLS * HIDDIM);
  k_scatter_agg<<<dim3(ECELL, HIDDIM / 128), 128, 0, stream>>>(ei, ei + ECELL, Hbuf, dinvc, AGG, HIDDIM);
  k_agg_epilogue<2><<<cdiv(NCELLS * HIDDIM, 256), 256, 0, stream>>>(
      AGG, Hbuf, dinvc, b1, nullptr, h2abf, NCELLS, HIDDIM);

  // GCN layer 2
  k_gemm_bf16<0,0><<<gemmNH, 256, 0, stream>>>(h2abf, W2Tbf, nullptr, Hbuf, nullptr,
                                               NCELLS, HIDDIM, HIDDIM);
  k_zero_f32<<<cdiv(NCELLS * HIDDIM, 256), 256, 0, stream>>>(AGG, NCELLS * HIDDIM);
  k_scatter_agg<<<dim3(ECELL, HIDDIM / 128), 128, 0, stream>>>(ei, ei + ECELL, Hbuf, dinvc, AGG, HIDDIM);
  k_agg_epilogue<2><<<cdiv(NCELLS * HIDDIM, 256), 256, 0, stream>>>(
      AGG, Hbuf, dinvc, b2, nullptr, h2bbf, NCELLS, HIDDIM);

  // h2 = leaky(h2b@W3 + b3) -> emb2 (fp32, normalized later) + h2bf
  k_gemm_bf16<2,0><<<gemmNH, 256, 0, stream>>>(h2bbf, W3Tbf, b3, emb2, h2bf,
                                               NCELLS, HIDDIM, HIDDIM);
  // h1 = relu(x0@Wm + bm) -> emb1 + h1bf
  k_gemm_bf16<1,0><<<gemmNH, 256, 0, stream>>>(x0bf, WmTbf, bm, emb1, h1bf,
                                               NCELLS, HIDDIM, NGENES);

  // gene branch: Hg = x0^T @ Wg ; aggregate ; relu -> xg
  k_gemm_bf16<0,0><<<dim3(1, cdiv(NGENES, BM)), 256, 0, stream>>>(
      x0Tbf, WgTbf, nullptr, Hbuf, nullptr, NGENES, OUTDIM, NCELLS);
  k_zero_f32<<<cdiv(NGENES * OUTDIM, 256), 256, 0, stream>>>(AGG, NGENES * OUTDIM);
  k_scatter_agg<<<dim3(EGENE, 1), 128, 0, stream>>>(eig, eig + EGENE, Hbuf, dinvg, AGG, OUTDIM);
  k_agg_epilogue<1><<<cdiv(NGENES * OUTDIM, 256), 256, 0, stream>>>(
      AGG, Hbuf, dinvg, bg, xgf, nullptr, NGENES, OUTDIM);
  k_transpose_cast<<<dim3(cdiv(OUTDIM, 32), cdiv(NGENES, 32)), dim3(32, 8), 0, stream>>>(
      xgf, xgTbf, NGENES, OUTDIM);

  // Wpxg = Wp @ xg [512x128], stored transposed bf16 [128x512]; bpxg = bp @ xg
  k_gemm_bf16<0,1><<<dim3(1, cdiv(HIDDIM, BM)), 256, 0, stream>>>(
      Wpbf, xgTbf, nullptr, nullptr, WpxgTbf, HIDDIM, OUTDIM, NGENES);
  k_bpxg<<<1, 128, 0, stream>>>(bp, xgf, bpxg);

  // z = l2norm(h @ Wpxg + bpxg)
  k_gemm_z_bf16<<<dim3(1, cdiv(NCELLS, BM)), 256, 0, stream>>>(h1bf, WpxgTbf, bpxg, z1, NCELLS, HIDDIM);
  k_gemm_z_bf16<<<dim3(1, cdiv(NCELLS, BM)), 256, 0, stream>>>(h2bf, WpxgTbf, bpxg, z2, NCELLS, HIDDIM);

  // emb = l2norm(h) in place
  k_l2norm_rows<<<cdiv(NCELLS * 32, 256), 256, 0, stream>>>(emb1, NCELLS, HIDDIM);
  k_l2norm_rows<<<cdiv(NCELLS * 32, 256), 256, 0, stream>>>(emb2, NCELLS, HIDDIM);
}
